// DotAtte_58823872086649
// MI455X (gfx1250) — matrix operand order repaired and checked by reading, not compiled, not run on hardware
//
#include <hip/hip_runtime.h>

typedef __attribute__((ext_vector_type(16))) __bf16 v16bf;
typedef __attribute__((ext_vector_type(8)))  __bf16 v8bf;
typedef __attribute__((ext_vector_type(8)))  float  v8f;
typedef __attribute__((ext_vector_type(4)))  float  v4f;

#define NB   8
#define SEQ  2048
#define DK   512
#define DV   512
#define BM   64          // query rows per block
#define BN   64          // keys per tile
#define QP   520         // Q LDS pitch (bf16) -> 1040B rows, conflict-free, 16B-mult
#define KP   520         // K LDS pitch (bf16)
#define VTP  72          // V^T LDS pitch (bf16) -> 144B rows, conflict-free, 16B-mult
#define SP   72          // score pitch (f32) -> 288B rows, 16B-mult
#define PP   144         // P pitch (bf16), aliased over Ss rows (2*SP)
#define SCALE 0.04419417382415922f   // 1/sqrt(512)

// A-matrix fragment (16x32 bf16): lanes 0-15 row M=lane hold K {k0..k0+7},{k0+16..k0+23};
// lanes 16-31 row M=lane-16 hold K {k0+8..k0+15},{k0+24..k0+31}.
__device__ __forceinline__ v16bf load_frag_a(const __bf16* row, int k0, int lane) {
  const int off = k0 + ((lane & 16) ? 8 : 0);
  v8bf lo = *(const v8bf*)(row + off);
  v8bf hi = *(const v8bf*)(row + off + 16);
  v16bf a;
  #pragma unroll
  for (int i = 0; i < 8; ++i) { a[i] = lo[i]; a[i + 8] = hi[i]; }
  return a;
}

// B-matrix fragment (32x16 bf16): lane<16 col N=lane holds K {k0..k0+7},{k0+16..k0+23};
// lane>=16 col N=lane-16 holds K {k0+8..k0+15},{k0+24..k0+31}.
__device__ __forceinline__ v16bf load_frag_b(const __bf16* row, int k0, int lane) {
  const int off = k0 + ((lane & 16) ? 8 : 0);
  v8bf lo = *(const v8bf*)(row + off);
  v8bf hi = *(const v8bf*)(row + off + 16);
  v16bf b;
  #pragma unroll
  for (int i = 0; i < 8; ++i) { b[i] = lo[i]; b[i + 8] = hi[i]; }
  return b;
}

__global__ __launch_bounds__(256, 1)
void fa_fwd_kernel(const float* __restrict__ Q, const float* __restrict__ K,
                   const float* __restrict__ V, const unsigned char* __restrict__ mask,
                   float* __restrict__ O)
{
  __shared__ __align__(16) __bf16 Qs[BM * QP];                 // 66,560 B
  __shared__ __align__(16) unsigned char KVraw[DV * VTP * 2];  // 73,728 B (>= BN*KP*2)
  __shared__ __align__(16) float  Ss[BM * SP];                 // 18,432 B (P aliased inside)
  __shared__ float alpha_s[BM];
  __shared__ float l_s[BM];

  __bf16* Kb = (__bf16*)KVraw;   // bf16 [BN][KP]   (phase 1)
  __bf16* Vt = (__bf16*)KVraw;   // bf16 [DV][VTP]  (phase 2, same buffer)
  __bf16* Ps = (__bf16*)Ss;      // bf16 [BM][PP]   (aliases score rows)

  const int tid  = threadIdx.x;
  const int wave = tid >> 5;
  const int lane = tid & 31;
  const int hi16 = lane >> 4;
  const int ln   = lane & 15;

  const int b  = blockIdx.x / (SEQ / BM);
  const int q0 = (blockIdx.x % (SEQ / BM)) * BM;

  const float* Qg = Q + ((size_t)b * SEQ + q0) * DK;
  const float* Kg = K + (size_t)b * SEQ * DK;
  const float* Vg = V + (size_t)b * SEQ * DV;
  const unsigned char* Mg = mask + (size_t)b * SEQ;

  // ---- Q tile (BM x DK) -> LDS, pre-scaled, bf16 ----
  for (int i = tid; i < BM * (DK / 4); i += 256) {
    const int row = i >> 7;              // DK/4 == 128
    const int c4  = (i & 127) << 2;
    v4f v = *(const v4f*)(Qg + row * DK + c4);
    __bf16* dst = &Qs[row * QP + c4];
    #pragma unroll
    for (int j = 0; j < 4; ++j) dst[j] = (__bf16)(v[j] * SCALE);
  }

  // Online-softmax state: each row owned by a fixed 4-lane group; value
  // replicated in all 4 lanes of the group.
  const int srow = (wave << 3) + (lane >> 2);  // this lane's softmax row
  const int chnk = lane & 3;                   // 16-col chunk within the row
  float m_run = -INFINITY;
  float l_run = 0.0f;

  v8f acc[4][4];
  {
    v8f z = {};
    #pragma unroll
    for (int mi = 0; mi < 4; ++mi)
      #pragma unroll
      for (int nj = 0; nj < 4; ++nj) acc[mi][nj] = z;
  }

  const int miw = wave & 3;            // phase-1 query sub-tile (16 rows)
  const int nsb = (wave >> 2) << 5;    // phase-1 key base: 0 or 32 (2 tiles/wave)
  const int dv0 = wave << 6;           // phase-2 DV base (64 cols per wave)

  for (int n0 = 0; n0 < SEQ; n0 += BN) {
    __syncthreads();   // protect KVraw / Ss / Ps from previous iteration

    // ---- K tile (BN x DK) -> bf16 LDS ----
    for (int i = tid; i < BN * (DK / 4); i += 256) {
      const int row = i >> 7;
      const int c4  = (i & 127) << 2;
      v4f v = *(const v4f*)(Kg + (size_t)(n0 + row) * DK + c4);
      if (n0 + BN < SEQ)
        __builtin_prefetch(Kg + (size_t)(n0 + BN + row) * DK + c4, 0, 1);
      __bf16* dst = &Kb[row * KP + c4];
      #pragma unroll
      for (int j = 0; j < 4; ++j) dst[j] = (__bf16)v[j];
    }
    __syncthreads();

    // ---- phase 1: S(BM x BN) = Q.K^T ; two 16x16 tiles per wave, shared A ----
    {
      v8f c0 = {}, c1 = {};
      const __bf16* arow  = Qs + ((miw << 4) + ln) * QP;
      const __bf16* brow0 = Kb + (nsb + ln) * KP;
      const __bf16* brow1 = Kb + (nsb + 16 + ln) * KP;
      for (int k0 = 0; k0 < DK; k0 += 32) {
        v16bf a  = load_frag_a(arow, k0, lane);
        v16bf b0 = load_frag_b(brow0, k0, lane);
        v16bf b1 = load_frag_b(brow1, k0, lane);
        c0 = __builtin_amdgcn_wmma_f32_16x16x32_bf16(false, a, false, b0,
                                                     (short)0, c0, false, false);
        c1 = __builtin_amdgcn_wmma_f32_16x16x32_bf16(false, a, false, b1,
                                                     (short)0, c1, false, false);
      }
      const int mr = (miw << 4) + (hi16 << 3);
      #pragma unroll
      for (int t = 0; t < 2; ++t) {
        const int  n  = nsb + (t << 4) + ln;
        const bool mk = (Mg[n0 + n] != 0);   // key-padding mask (per column)
        const v8f& c  = t ? c1 : c0;
        #pragma unroll
        for (int r = 0; r < 8; ++r)
          Ss[(mr + r) * SP + n] = mk ? -INFINITY : c[r];
      }
    }
    __syncthreads();

    // ---- online softmax: 4 lanes per row, shfl reductions, all waves busy ----
    {
      const float* sp_ = &Ss[srow * SP + (chnk << 4)];
      float sc[16];
      #pragma unroll
      for (int j = 0; j < 16; ++j) sc[j] = sp_[j];

      float mx = m_run;
      #pragma unroll
      for (int j = 0; j < 16; ++j) mx = fmaxf(mx, sc[j]);
      mx = fmaxf(mx, __shfl_xor(mx, 1, 32));
      mx = fmaxf(mx, __shfl_xor(mx, 2, 32));

      const float alpha = (mx == -INFINITY) ? 1.0f : __expf(m_run - mx);
      float lsum = 0.0f;
      __bf16 pb[16];
      #pragma unroll
      for (int j = 0; j < 16; ++j) {
        const float p = (mx == -INFINITY) ? 0.0f : __expf(sc[j] - mx);
        pb[j] = (__bf16)p;
        lsum += p;
      }
      lsum += __shfl_xor(lsum, 1, 32);
      lsum += __shfl_xor(lsum, 2, 32);

      // P writeback (aliases the score rows; all reads above are in registers)
      __bf16* pp_ = &Ps[srow * PP + (chnk << 4)];
      #pragma unroll
      for (int j = 0; j < 16; ++j) pp_[j] = pb[j];

      m_run = mx;
      l_run = l_run * alpha + lsum;
      if (chnk == 0) alpha_s[srow] = alpha;
    }
    __syncthreads();

    // ---- rescale accumulators by alpha(row) ----
    #pragma unroll
    for (int mi = 0; mi < 4; ++mi) {
      float al[8];
      #pragma unroll
      for (int r = 0; r < 8; ++r)
        al[r] = alpha_s[(mi << 4) + (hi16 << 3) + r];
      #pragma unroll
      for (int nj = 0; nj < 4; ++nj)
        #pragma unroll
        for (int r = 0; r < 8; ++r)
          acc[mi][nj][r] *= al[r];
    }

    // ---- V tile (BN x DV) -> transposed bf16 LDS (reuses K buffer) ----
    for (int i = tid; i < BN * (DV / 4); i += 256) {
      const int row = i >> 7;
      const int c4  = (i & 127) << 2;
      v4f v = *(const v4f*)(Vg + (size_t)(n0 + row) * DV + c4);
      if (n0 + BN < SEQ)
        __builtin_prefetch(Vg + (size_t)(n0 + BN + row) * DV + c4, 0, 1);
      #pragma unroll
      for (int j = 0; j < 4; ++j) Vt[(c4 + j) * VTP + row] = (__bf16)v[j];
    }
    __syncthreads();

    // ---- phase 2: acc += P(BM x BN) . V(BN x DV), 16 tiles per wave ----
    #pragma unroll
    for (int ks = 0; ks < 2; ++ks) {
      const int k0 = ks << 5;
      v16bf pa[4];
      #pragma unroll
      for (int mi = 0; mi < 4; ++mi)
        pa[mi] = load_frag_a(Ps + ((mi << 4) + ln) * PP, k0, lane);
      #pragma unroll
      for (int nj = 0; nj < 4; ++nj) {
        v16bf vb = load_frag_b(Vt + (dv0 + (nj << 4) + ln) * VTP, k0, lane);
        #pragma unroll
        for (int mi = 0; mi < 4; ++mi)
          acc[mi][nj] = __builtin_amdgcn_wmma_f32_16x16x32_bf16(
              false, pa[mi], false, vb, (short)0, acc[mi][nj], false, false);
      }
    }
  }

  if (chnk == 0) l_s[srow] = l_run;
  __syncthreads();

  float* Og = O + ((size_t)b * SEQ + q0) * DV;
  #pragma unroll
  for (int mi = 0; mi < 4; ++mi) {
    float linv[8];
    #pragma unroll
    for (int r = 0; r < 8; ++r)
      linv[r] = 1.0f / l_s[(mi << 4) + (hi16 << 3) + r];
    const int mr = (mi << 4) + (hi16 << 3);
    #pragma unroll
    for (int nj = 0; nj < 4; ++nj) {
      const int n = dv0 + (nj << 4) + ln;
      #pragma unroll
      for (int r = 0; r < 8; ++r)
        Og[(size_t)(mr + r) * DV + n] = acc[mi][nj][r] * linv[r];
    }
  }
}

extern "C" void kernel_launch(void* const* d_in, const int* in_sizes, int n_in,
                              void* d_out, int out_size, void* d_ws, size_t ws_size,
                              hipStream_t stream) {
  (void)in_sizes; (void)n_in; (void)out_size; (void)d_ws; (void)ws_size;
  const float* Q = (const float*)d_in[0];
  const float* K = (const float*)d_in[1];
  const float* V = (const float*)d_in[2];
  const unsigned char* mask = (const unsigned char*)d_in[3];
  float* O = (float*)d_out;

  dim3 grid(NB * (SEQ / BM));   // 256 blocks
  dim3 block(256);              // 8 wave32 waves
  hipLaunchKernelGGL(fa_fwd_kernel, grid, block, 0, stream, Q, K, V, mask, O);
}
